// SequenceModel_62818191671709
// MI455X (gfx1250) — compile-verified, matrix operand order checked by reading, not run on hardware
//
#include <hip/hip_runtime.h>

typedef __attribute__((ext_vector_type(16))) _Float16 v16h;
typedef __attribute__((ext_vector_type(8)))  _Float16 v8h;
typedef __attribute__((ext_vector_type(8)))  float    v8f;

#define II 10
#define HH 64

#if __has_builtin(__builtin_amdgcn_tanhf)
#define TANHF(x) __builtin_amdgcn_tanhf(x)
#else
#define TANHF(x) tanhf(x)
#endif

// One wave32 per 16-row batch tile. Weights live in B-fragment registers for
// the entire T=2048 recurrence; hidden state round-trips through LDS (f16),
// which performs the D-fragment -> A-fragment relayout. All per-step control
// (x K-padding, variable-length freeze) is branch-free (v_cndmask, not exec).
__global__ __launch_bounds__(32) void rnn_wmma_kernel(
    const float* __restrict__ input_seq,   // [B, T, I]
    const int*   __restrict__ seq_lengths, // [B]
    const float* __restrict__ W_xh,        // [H, I]
    const float* __restrict__ W_hh,        // [H, H]
    const float* __restrict__ b_hh,        // [H]
    const float* __restrict__ W_out,       // [O=1, H]
    const float* __restrict__ b_out,       // [O=1]
    float* __restrict__ out,               // [B, 1]
    int T)
{
    __shared__ __align__(16) _Float16 h16[16 * HH];

    const int lane = threadIdx.x & 31;
    const int half = lane >> 4;   // lane group (0: lanes 0-15, 1: lanes 16-31)
    const int mrow = lane & 15;
    const int row0 = blockIdx.x * 16;

    // h0 = 0
    for (int i = lane; i < 16 * HH; i += 32) h16[i] = (_Float16)0.f;

    // ---- weight B-fragments, built once ----
    // element e, lane L: Ncol = n*16 + (L&15); K = (e>>3)*16 + (L>>4)*8 + (e&7)
    v16h bxh[4];     // for x @ W_xh^T, K padded 10 -> 32
    v16h bhh[2][4];  // for h @ W_hh^T, K = 64 as two chunks of 32
    v8f  biasf[4];   // b_hh broadcast down rows (C operand of first WMMA)
#pragma unroll
    for (int n = 0; n < 4; ++n) {
        const int ncol = n * 16 + mrow;
#pragma unroll
        for (int e = 0; e < 16; ++e) {
            const int K = (e >> 3) * 16 + half * 8 + (e & 7);
            bxh[n][e]    = (K < II) ? (_Float16)W_xh[ncol * II + K] : (_Float16)0.f;
            bhh[0][n][e] = (_Float16)W_hh[ncol * HH + K];
            bhh[1][n][e] = (_Float16)W_hh[ncol * HH + 32 + K];
        }
        const float bv = b_hh[ncol];
#pragma unroll
        for (int r = 0; r < 8; ++r) biasf[n][r] = bv;
    }

    // seq lengths for the 8 rows this lane's C/D registers cover (M = r + 8*half)
    int lens[8];
#pragma unroll
    for (int r = 0; r < 8; ++r) lens[r] = seq_lengths[row0 + r + 8 * half];

    // persistent hidden tile in D-fragment layout (select target for freezing)
    _Float16 hreg[4][8];
#pragma unroll
    for (int n = 0; n < 4; ++n)
#pragma unroll
        for (int r = 0; r < 8; ++r) hreg[n][r] = (_Float16)0.f;

    const float* xrow = input_seq + (size_t)(row0 + mrow) * T * II;

    __syncthreads();

    for (int t = 0; t < T; ++t) {
        // ---- A fragment for x_t (16x32, K >= 10 zero) ----
        // Valid K for this lane: half*8 + {0..7}; pairs are 8-byte aligned
        // (row stride 40 B). Clamp index + cndmask select: no exec branching.
        const float* xp = xrow + (size_t)t * II;
        if (t + 1 < T) __builtin_prefetch(xp + II, 0, 0);
        const float2* xp2 = (const float2*)xp;
        v16h ax;
#pragma unroll
        for (int p = 0; p < 4; ++p) {
            const int K = half * 8 + 2 * p;
            const float2 v = xp2[(K < II ? K : 0) >> 1];
            const bool ok = (K < II);
            ax[2 * p]     = ok ? (_Float16)v.x : (_Float16)0.f;
            ax[2 * p + 1] = ok ? (_Float16)v.y : (_Float16)0.f;
        }
#pragma unroll
        for (int e = 8; e < 16; ++e) ax[e] = (_Float16)0.f;  // K = 16..31 pad

        // ---- A fragments for h_{t-1}: two 16x32 chunks, 2x b128 DS loads each ----
        const v8h* hp = (const v8h*)h16;
        v8h a0lo = hp[(mrow * HH +      half * 8) >> 3];
        v8h a0hi = hp[(mrow * HH + 16 + half * 8) >> 3];
        v8h a1lo = hp[(mrow * HH + 32 +      half * 8) >> 3];
        v8h a1hi = hp[(mrow * HH + 32 + 16 + half * 8) >> 3];
        v16h ah0 = __builtin_shufflevector(a0lo, a0hi, 0,1,2,3,4,5,6,7,8,9,10,11,12,13,14,15);
        v16h ah1 = __builtin_shufflevector(a1lo, a1hi, 0,1,2,3,4,5,6,7,8,9,10,11,12,13,14,15);

        // ---- 12 WMMAs: acc = bias + x Wxh^T + h Whh^T, per 16-col tile ----
        v8f acc[4];
#pragma unroll
        for (int n = 0; n < 4; ++n) {
            acc[n] = __builtin_amdgcn_wmma_f32_16x16x32_f16(false, ax,  false, bxh[n],
                                                            (short)0, biasf[n], false, false);
            acc[n] = __builtin_amdgcn_wmma_f32_16x16x32_f16(false, ah0, false, bhh[0][n],
                                                            (short)0, acc[n],   false, false);
            acc[n] = __builtin_amdgcn_wmma_f32_16x16x32_f16(false, ah1, false, bhh[1][n],
                                                            (short)0, acc[n],   false, false);
        }

        // ---- tanh + length-masked hidden update (cndmask, unconditional store) ----
#pragma unroll
        for (int n = 0; n < 4; ++n) {
            const int ncol = n * 16 + mrow;
#pragma unroll
            for (int r = 0; r < 8; ++r) {
                const _Float16 hn = (_Float16)TANHF(acc[n][r]);
                hreg[n][r] = (t < lens[r]) ? hn : hreg[n][r];
                h16[(r + 8 * half) * HH + ncol] = hreg[n][r];
            }
        }
        __syncthreads(); // single-wave WG: nop barrier; DS RAW handled by dscnt
    }

    // ---- output projection: out[b] = h[b,:] . W_out[0,:] + b_out ----
    if (half == 0) {
        float s = b_out[0];
#pragma unroll 8
        for (int k = 0; k < HH; ++k)
            s += (float)h16[mrow * HH + k] * W_out[k];
        out[row0 + mrow] = s;
    }
}

extern "C" void kernel_launch(void* const* d_in, const int* in_sizes, int n_in,
                              void* d_out, int out_size, void* d_ws, size_t ws_size,
                              hipStream_t stream) {
    const float* input_seq   = (const float*)d_in[0];
    const int*   seq_lengths = (const int*)d_in[1];
    const float* W_xh        = (const float*)d_in[2];
    const float* W_hh        = (const float*)d_in[3];
    const float* b_hh        = (const float*)d_in[4];
    const float* W_out       = (const float*)d_in[5];
    const float* b_out       = (const float*)d_in[6];
    float* out = (float*)d_out;

    const int Btot = in_sizes[1];                 // 2048
    const int T    = in_sizes[0] / (Btot * II);   // 2048

    dim3 grid(Btot / 16), block(32);
    hipLaunchKernelGGL(rnn_wmma_kernel, grid, block, 0, stream,
                       input_seq, seq_lengths, W_xh, W_hh, b_hh, W_out, b_out, out, T);
}
